// GLA_54589034332317
// MI455X (gfx1250) — matrix operand order repaired and checked by reading, not compiled, not run on hardware
//
#include <hip/hip_runtime.h>
#include <math.h>

// ---------------- problem constants (from reference) ----------------
#define N_B    2
#define CH     64
#define HH     96
#define WW     96
#define LL     (HH*WW)        // 9216
#define NHH    4
#define CHUNKK 144
#define CMM    16
#define HB     64
#define NCHK   (LL/CHUNKK)    // 64 chunks
#define SORTN  (NHH*LL)       // 36864
#define NBINS  256
#define NEPS   5e-5f

#define ATTN_THREADS 288      // 9 waves: one 16-row i-tile per wave (no spills)

typedef __attribute__((ext_vector_type(16))) _Float16 v16h;
typedef __attribute__((ext_vector_type(8)))  float    v8f;

// ---------------- WMMA helpers (CDNA5 gfx1250, wave32) ----------------
__device__ __forceinline__ v8f wmma16(v16h a, v16h b, v8f c) {
  // v_wmma_f32_16x16x32_f16 : D = A(16x32) x B(32x16) + C(16x16 f32)
  return __builtin_amdgcn_wmma_f32_16x16x32_f16(
      /*neg_a=*/false, a, /*neg_b=*/false, b,
      /*c_mod=*/(short)0, c, /*reuse_a=*/false, /*reuse_b=*/false);
}

// A fragment: A[m][k], m = lane&15; per ISA 16-bit A 16x32 layout:
// half=0 -> K in {0..7, 16..23}; half=1 -> K in {8..15, 24..31}
// (two contiguous 8-half runs per lane -> ds_load_b128 friendly)
__device__ __forceinline__ v16h load_a16(const _Float16* base, int rs, int row0, int k0) {
  int lane = threadIdx.x & 31; int hw = lane >> 4; int m = lane & 15;
  v16h a;
#pragma unroll
  for (int e = 0; e < 16; ++e) {
    int k = ((e < 8) ? e : e + 8) + hw * 8;
    a[e] = base[(size_t)(row0 + m) * rs + k0 + k];
  }
  return a;
}

// B fragment from a ROW-PER-N (transposed) store: B[k][n] = base[(col0+n)*rs + k0 + k]
// per-lane elements k = {0..7,16..23} + 8*hw (same order as A) -> 2x ds_load_b128
__device__ __forceinline__ v16h load_bT16(const _Float16* base, int rs, int k0, int col0) {
  int lane = threadIdx.x & 31; int hw = lane >> 4; int nn = lane & 15;
  v16h b;
#pragma unroll
  for (int e = 0; e < 16; ++e) {
    b[e] = base[(size_t)(col0 + nn) * rs + k0 + ((e < 8) ? e : e + 8) + hw * 8];
  }
  return b;
}

// ---------------- K1: fused 3x3 convs (xe f32 16ch; ye/fe f16 64ch) ----------------
__global__ void gla_conv_kernel(const float* __restrict__ x,
                                const float* __restrict__ wm, const float* __restrict__ bm,
                                const float* __restrict__ wa, const float* __restrict__ ba,
                                const float* __restrict__ wf, const float* __restrict__ bf,
                                float* __restrict__ xe, _Float16* __restrict__ yeh,
                                _Float16* __restrict__ feh) {
  int t = blockIdx.x * blockDim.x + threadIdx.x;
  const int total = N_B * LL * 144;
  if (t >= total) return;
  int oc = t % 144;
  int l  = (t / 144) % LL;
  int n  = t / (144 * LL);
  int y = l / WW, xc = l % WW;
  const float* W; float bias; int o;
  if (oc < CMM)            { o = oc;            W = wm; bias = bm[o]; }
  else if (oc < CMM + CH)  { o = oc - CMM;      W = wa; bias = ba[o]; }
  else                     { o = oc - CMM - CH; W = wf; bias = bf[o]; }
  float acc = bias;
  for (int i = 0; i < CH; ++i) {
    const float* xp = x + ((size_t)(n * CH + i) * HH) * WW;
    const float* wp = W + ((size_t)(o * CH + i)) * 9;
#pragma unroll
    for (int ky = 0; ky < 3; ++ky) {
      int yy = y + ky - 1; if (yy < 0 || yy >= HH) continue;
#pragma unroll
      for (int kx = 0; kx < 3; ++kx) {
        int xx = xc + kx - 1; if (xx < 0 || xx >= WW) continue;
        acc += xp[yy * WW + xx] * wp[ky * 3 + kx];
      }
    }
  }
  acc = fmaxf(acc, 0.f);
  if (oc < CMM)           xe [((size_t)(n * LL + l)) * CMM + o] = acc;
  else if (oc < CMM + CH) yeh[((size_t)(n * LL + l)) * CH  + o] = (_Float16)acc;
  else                    feh[((size_t)(n * LL + l)) * CH  + o] = (_Float16)acc;
}

// ---------------- K2: LSH codes (argmax of rotated dirs) ----------------
__global__ void gla_codes_kernel(const float* __restrict__ xe,
                                 const float* __restrict__ rot,
                                 int* __restrict__ codes) {
  int t = blockIdx.x * blockDim.x + threadIdx.x;
  const int total = N_B * NHH * LL;
  if (t >= total) return;
  int l = t % LL;
  int h = (t / LL) % NHH;
  int n = t / (LL * NHH);
  float v[CMM];
#pragma unroll
  for (int f = 0; f < CMM; ++f) v[f] = xe[((size_t)n * LL + l) * CMM + f];
  float best = -INFINITY; int bi = 0;
  for (int i = 0; i < HB; ++i) {
    float s = 0.f;
#pragma unroll
    for (int f = 0; f < CMM; ++f) s += v[f] * rot[(size_t)(f * NHH + h) * HB + i];
    if (s > best) { best = s; bi = i; }  // first max wins (stable like jnp.argmax)
  }
  codes[(size_t)n * SORTN + h * LL + l] = bi + h * HB;
}

// ---------------- K3/K4/K5/K6: stable counting sort ----------------
__global__ void gla_zero_kernel(int* __restrict__ p, int n) {
  int t = blockIdx.x * blockDim.x + threadIdx.x;
  if (t < n) p[t] = 0;
}

__global__ void gla_hist_kernel(const int* __restrict__ codes, int* __restrict__ counts) {
  int t = blockIdx.x * blockDim.x + threadIdx.x;
  if (t >= N_B * SORTN) return;
  int n = t / SORTN;
  atomicAdd(&counts[n * NBINS + codes[t]], 1);
}

__global__ void gla_scan_kernel(const int* __restrict__ counts, int* __restrict__ bin_start) {
  int n = threadIdx.x;
  if (n < N_B) {
    int base = 0;
    for (int b = 0; b < NBINS; ++b) {
      bin_start[n * NBINS + b] = base;
      base += counts[n * NBINS + b];
    }
  }
}

// one wave per (bin, n): stable pack via ballot/popc
__global__ void gla_pack_kernel(const int* __restrict__ codes,
                                const int* __restrict__ bin_start,
                                int* __restrict__ modp, int* __restrict__ undo) {
  int bin  = blockIdx.x;
  int n    = blockIdx.y;
  int lane = threadIdx.x;  // 0..31
  int base = bin_start[n * NBINS + bin];
  const int* cp = codes + (size_t)n * SORTN;
  for (int e0 = 0; e0 < SORTN; e0 += 32) {       // SORTN % 32 == 0
    int e = e0 + lane;
    bool m = (cp[e] == bin);
    unsigned mask = (unsigned)__ballot(m);
    if (m) {
      int rank = __popc(mask & ((1u << lane) - 1u));
      int pos  = base + rank;
      modp[(size_t)n * SORTN + pos] = e % LL;
      undo[(size_t)n * SORTN + e]   = pos;
    }
    base += __popc(mask);
  }
}

// ---------------- K7: flash-style chunked attention block ----------------
// one 288-thread (9-wave) workgroup per (n, h, chunk); each wave owns one
// 16-row i-tile, so the O accumulator stays in 32 VGPRs (no spills).
#define FST  72   // stride of fa staging rows (64 + pad)
#define BSTT 168  // stride of transposed [xm|hid] panel rows (160 + pad) -> 336B
#define YST  40   // stride of transposed ya panel rows (32 + pad) -> 80B
#define PST  40   // stride of P buffer rows

__global__ void __launch_bounds__(ATTN_THREADS, 1)
gla_attn_kernel(const float* __restrict__ xe, const _Float16* __restrict__ yeh,
                const _Float16* __restrict__ feh, const int* __restrict__ modp,
                const float* __restrict__ w1g, const float* __restrict__ b1g,
                const float* __restrict__ w2g, const float* __restrict__ b2g,
                float* __restrict__ ret, float* __restrict__ bscp) {
  const int k = blockIdx.x;   // chunk
  const int h = blockIdx.y;   // head
  const int n = blockIdx.z;   // batch
  const int tid  = threadIdx.x;
  const int wave = tid >> 5;  // 0..8  == i-tile index
  const int lane = tid & 31;
  const int hw   = lane >> 4;
  const int ln   = lane & 15;
  const int ib   = wave * 16; // rows [ib, ib+16) owned by this wave

  extern __shared__ unsigned char smem[];
  float* s_b1 = (float*)smem;            // 144
  float* s_b2 = s_b1 + CHUNKK;           // 144
  float* s_m  = s_b2 + CHUNKK;           // 144 running max
  float* s_s  = s_m  + CHUNKK;           // 144 running sum
  int*   s_lj = (int*)(s_s + CHUNKK);    // 32 gathered row ids
  _Float16* sA   = (_Float16*)(s_lj + 32);   // [144][160] = [xs | fc2_w]  (A panel)
  _Float16* sW1  = sA  + CHUNKK * 160;       // [144][64]  fc1_w           (B via bT)
  _Float16* sBt  = sW1 + CHUNKK * CH;        // [32][BSTT] transposed [xm|hid] panel
  _Float16* sFA  = sBt + 32 * BSTT;          // [32][FST]  fa j-slice      (A panel)
  _Float16* sYAt = sFA + 32 * FST;           // [64][YST]  transposed ya panel
  _Float16* sP   = sYAt + CH * YST;          // [9 bufs][16][PST] score tiles

  // ---- prologue: A panel = [xs | fc2_w], weights, biases, flash state ----
  for (int idx = tid; idx < CHUNKK * CMM; idx += ATTN_THREADS) {
    int i = idx >> 4, e = idx & 15;
    int l = modp[(size_t)n * SORTN + h * LL + k * CHUNKK + i];
    sA[i * 160 + e] = (_Float16)xe[((size_t)n * LL + l) * CMM + e];
  }
  for (int idx = tid; idx < CHUNKK * CHUNKK; idx += ATTN_THREADS) {
    int i = idx / CHUNKK, e = idx % CHUNKK;
    sA[i * 160 + 16 + e] = (_Float16)w2g[(size_t)i * CHUNKK + e];
  }
  for (int idx = tid; idx < CHUNKK * CH; idx += ATTN_THREADS)
    sW1[idx] = (_Float16)w1g[idx];
  for (int idx = tid; idx < CHUNKK; idx += ATTN_THREADS) {
    s_b1[idx] = b1g[idx];
    s_b2[idx] = b2g[idx];
    s_m[idx]  = -INFINITY;
    s_s[idx]  = 0.f;
  }

  v8f Oacc[4];
#pragma unroll
  for (int nt = 0; nt < 4; ++nt)
#pragma unroll
    for (int v = 0; v < 8; ++v) Oacc[nt][v] = 0.f;

  // ---- stream 432 key columns in 14 steps of 32 (last step: 16 + mask) ----
  for (int js = 0; js < 14; ++js) {
    __syncthreads();  // previous step's consumers done
    const int jbase = js * 32;
    const int ncols = (js == 13) ? 16 : 32;
    const bool full = (js < 13);

    if (tid < 32) {
      int jp = tid, l = -1;
      if (jp < ncols) {
        int jg  = jbase + jp;
        int seg = jg / CHUNKK, r = jg % CHUNKK;
        int ck  = (seg == 0) ? k : (seg == 1) ? (k + NCHK - 1) % NCHK : (k + 1) % NCHK;
        l = modp[(size_t)n * SORTN + h * LL + ck * CHUNKK + r];
      }
      s_lj[jp] = l;
    }
    __syncthreads();

    // stage fa (row-major, A-operand) and ya (transposed, B-operand) slices
    for (int idx = tid; idx < 32 * CH; idx += ATTN_THREADS) {
      int jp = idx >> 6, c = idx & 63;
      int l = s_lj[jp];
      sFA[jp * FST + c]  = (l >= 0) ? feh[((size_t)n * LL + l) * CH + c] : (_Float16)0.f;
      sYAt[c * YST + jp] = (l >= 0) ? yeh[((size_t)n * LL + l) * CH + c] : (_Float16)0.f;
    }
    // transposed panel rows: sBt[j'][0..15] = xmatch (normalized xe row)
    if (tid < 32) {
      int jp = tid, l = s_lj[jp];
      if (l >= 0) {
        float vv[CMM]; float ssq = 0.f;
#pragma unroll
        for (int e = 0; e < CMM; ++e) {
          vv[e] = xe[((size_t)n * LL + l) * CMM + e];
          ssq += vv[e] * vv[e];
        }
        float inv = 1.f / fmaxf(sqrtf(ssq), NEPS);
#pragma unroll
        for (int e = 0; e < CMM; ++e) sBt[jp * BSTT + e] = (_Float16)(vv[e] * inv);
      } else {
#pragma unroll
        for (int e = 0; e < CMM; ++e) sBt[jp * BSTT + e] = (_Float16)0.f;
      }
    }
    __syncthreads();

    // GEMM1 (cooperative): hid tile = relu(fa . fc1^T + b1) -> sBt[j'][16..159]
    for (int job = wave; job < 18; job += 9) {
      int mt = job & 1, nt = job >> 1;
      float bb = s_b1[nt * 16 + ln];
      v8f acc;
#pragma unroll
      for (int v = 0; v < 8; ++v) acc[v] = bb;
#pragma unroll
      for (int ks = 0; ks < 2; ++ks) {
        v16h a = load_a16(sFA, FST, mt * 16, ks * 32);
        v16h b = load_bT16(sW1, CH, ks * 32, nt * 16);
        acc = wmma16(a, b, acc);
      }
#pragma unroll
      for (int v = 0; v < 8; ++v) {
        int e  = nt * 16 + ln;
        int jr = mt * 16 + v + hw * 8;
        sBt[jr * BSTT + 16 + e] = (_Float16)fmaxf(acc[v], 0.f);
      }
    }
    __syncthreads();

    // GEMM2 + online softmax: raw = [xs|w2] . [xm|hid]^T + b2
    {
      v8f acc0, acc1;
#pragma unroll
      for (int v = 0; v < 8; ++v) {
        float bb = s_b2[ib + v + hw * 8];
        acc0[v] = bb; acc1[v] = bb;
      }
#pragma unroll
      for (int ks = 0; ks < 5; ++ks) {
        v16h a  = load_a16(sA, 160, ib, ks * 32);
        v16h b0 = load_bT16(sBt, BSTT, ks * 32, 0);
        v16h b1 = load_bT16(sBt, BSTT, ks * 32, 16);
        acc0 = wmma16(a, b0, acc0);
        acc1 = wmma16(a, b1, acc1);
      }
      float p0[8], p1[8], scl[8];
#pragma unroll
      for (int v = 0; v < 8; ++v) {
        float t = acc0[v];
        if (full) t = fmaxf(t, acc1[v]);
        t = fmaxf(t, __shfl_xor(t, 1));
        t = fmaxf(t, __shfl_xor(t, 2));
        t = fmaxf(t, __shfl_xor(t, 4));
        t = fmaxf(t, __shfl_xor(t, 8));
        int row = ib + v + hw * 8;
        float mold = s_m[row];
        float mnew = fmaxf(mold, t);
        float sc = (mold == -INFINITY) ? 0.f : expf(mold - mnew);
        float e0 = expf(acc0[v] - mnew);
        float e1 = full ? expf(acc1[v] - mnew) : 0.f;
        float rs = e0 + e1;
        rs += __shfl_xor(rs, 1);
        rs += __shfl_xor(rs, 2);
        rs += __shfl_xor(rs, 4);
        rs += __shfl_xor(rs, 8);
        float snew = s_s[row] * sc + rs;
        if (ln == 0) { s_m[row] = mnew; s_s[row] = snew; }
        p0[v] = e0; p1[v] = e1; scl[v] = sc;
      }
#pragma unroll
      for (int nt = 0; nt < 4; ++nt)
#pragma unroll
        for (int v = 0; v < 8; ++v) Oacc[nt][v] *= scl[v];
      // transpose score tile to A-fragment layout via per-wave LDS buffer
      _Float16* Pw = sP + (size_t)wave * 16 * PST;
#pragma unroll
      for (int v = 0; v < 8; ++v) {
        Pw[(v + hw * 8) * PST + ln]      = (_Float16)p0[v];
        Pw[(v + hw * 8) * PST + 16 + ln] = (_Float16)p1[v];
      }
    }
    __syncthreads();

    // GEMM3: O += P . ya   (K = 32 j-columns; masked tail contributes zeros)
    {
      const _Float16* Pw = sP + (size_t)wave * 16 * PST;
      v16h aP = load_a16(Pw, PST, 0, 0);
#pragma unroll
      for (int nt = 0; nt < 4; ++nt) {
        v16h bY = load_bT16(sYAt, YST, 0, nt * 16);
        Oacc[nt] = wmma16(aP, bY, Oacc[nt]);
      }
    }
  }

  // ---- epilogue: normalize, write ret_sorted + bsc_sorted ----
  __syncthreads();
#pragma unroll
  for (int nt = 0; nt < 4; ++nt) {
#pragma unroll
    for (int v = 0; v < 8; ++v) {
      int row = ib + v + hw * 8;
      float val = Oacc[nt][v] / s_s[row];
      size_t p = (size_t)h * LL + (size_t)k * CHUNKK + row;
      ret[((size_t)n * SORTN + p) * CH + nt * 16 + ln] = val;
    }
  }
  if (ln == 0) {
#pragma unroll
    for (int v = 0; v < 8; ++v) {
      int row = ib + v + hw * 8;
      size_t p = (size_t)h * LL + (size_t)k * CHUNKK + row;
      bscp[(size_t)n * SORTN + p] = s_m[row] + logf(s_s[row]);
    }
  }
}

// dynamic LDS size for gla_attn_kernel (~97 KB < 320 KB/WGP)
static constexpr size_t GLA_ATTN_SMEM =
    (4 * CHUNKK) * sizeof(float) + 32 * sizeof(int) +
    ((size_t)CHUNKK * 160 + (size_t)CHUNKK * CH + 32 * BSTT +
     32 * FST + CH * YST + 9 * 16 * PST) * sizeof(_Float16);

// ---------------- K8: undo sort, head softmax, residual ----------------
__global__ void gla_final_kernel(const float* __restrict__ x,
                                 const float* __restrict__ ret,
                                 const float* __restrict__ bscp,
                                 const int* __restrict__ undo,
                                 float* __restrict__ out) {
  int t = blockIdx.x * blockDim.x + threadIdx.x;
  if (t >= N_B * LL) return;
  int l = t % LL, n = t / LL;
  int pos[NHH]; float b[NHH];
  float mx = -INFINITY;
#pragma unroll
  for (int h2 = 0; h2 < NHH; ++h2) {
    pos[h2] = undo[(size_t)n * SORTN + h2 * LL + l];
    b[h2]   = bscp[(size_t)n * SORTN + pos[h2]];
    mx = fmaxf(mx, b[h2]);
  }
  float pw[NHH]; float den = 0.f;
#pragma unroll
  for (int h2 = 0; h2 < NHH; ++h2) { pw[h2] = expf(b[h2] - mx); den += pw[h2]; }
  float inv = 1.f / den;
  for (int c = 0; c < CH; ++c) {
    float s = 0.f;
#pragma unroll
    for (int h2 = 0; h2 < NHH; ++h2)
      s += pw[h2] * ret[((size_t)n * SORTN + pos[h2]) * CH + c];
    size_t oi = ((size_t)(n * CH + c)) * LL + l;
    out[oi] = s * inv + x[oi];  // RES_SCALE == 1.0
  }
}

// ---------------- host launcher ----------------
extern "C" void kernel_launch(void* const* d_in, const int* in_sizes, int n_in,
                              void* d_out, int out_size, void* d_ws, size_t ws_size,
                              hipStream_t stream) {
  (void)in_sizes; (void)n_in; (void)out_size; (void)ws_size;
  const float* x     = (const float*)d_in[0];
  const float* rot   = (const float*)d_in[1];
  const float* wm    = (const float*)d_in[2];
  const float* bm    = (const float*)d_in[3];
  const float* wa    = (const float*)d_in[4];
  const float* ba    = (const float*)d_in[5];
  const float* wf    = (const float*)d_in[6];
  const float* bf    = (const float*)d_in[7];
  const float* fc1_w = (const float*)d_in[8];
  const float* fc1_b = (const float*)d_in[9];
  const float* fc2_w = (const float*)d_in[10];
  const float* fc2_b = (const float*)d_in[11];
  float* out = (float*)d_out;

  // carve workspace
  char* ws = (char*)d_ws;
  size_t off = 0;
  auto carve = [&](size_t bytes) -> char* {
    char* p = ws + off;
    off = (off + bytes + 255) & ~(size_t)255;
    return p;
  };
  float*    xe  = (float*)   carve((size_t)N_B * LL * CMM * sizeof(float));
  _Float16* yeh = (_Float16*)carve((size_t)N_B * LL * CH * sizeof(_Float16));
  _Float16* feh = (_Float16*)carve((size_t)N_B * LL * CH * sizeof(_Float16));
  float*    ret = (float*)   carve((size_t)N_B * SORTN * CH * sizeof(float));
  float*    bsc = (float*)   carve((size_t)N_B * SORTN * sizeof(float));
  int* codes     = (int*)carve((size_t)N_B * SORTN * sizeof(int));
  int* counts    = (int*)carve((size_t)N_B * NBINS * sizeof(int));
  int* bin_start = (int*)carve((size_t)N_B * NBINS * sizeof(int));
  int* modp      = (int*)carve((size_t)N_B * SORTN * sizeof(int));
  int* undo      = (int*)carve((size_t)N_B * SORTN * sizeof(int));

  // K1: fused convs
  {
    int total = N_B * LL * 144;
    gla_conv_kernel<<<(total + 255) / 256, 256, 0, stream>>>(
        x, wm, bm, wa, ba, wf, bf, xe, yeh, feh);
  }
  // K2: codes
  {
    int total = N_B * NHH * LL;
    gla_codes_kernel<<<(total + 255) / 256, 256, 0, stream>>>(xe, rot, codes);
  }
  // K3..K6: stable counting sort -> modp / undo
  gla_zero_kernel<<<(N_B * NBINS + 255) / 256, 256, 0, stream>>>(counts, N_B * NBINS);
  gla_hist_kernel<<<(N_B * SORTN + 255) / 256, 256, 0, stream>>>(codes, counts);
  gla_scan_kernel<<<1, 64, 0, stream>>>(counts, bin_start);
  gla_pack_kernel<<<dim3(NBINS, N_B), 32, 0, stream>>>(codes, bin_start, modp, undo);
  // K7: flash attention blocks (WMMA)
  gla_attn_kernel<<<dim3(NCHK, NHH, N_B), ATTN_THREADS, GLA_ATTN_SMEM, stream>>>(
      xe, yeh, feh, modp, fc1_w, fc1_b, fc2_w, fc2_b, ret, bsc);
  // K8: undo + head softmax + residual
  gla_final_kernel<<<(N_B * LL + 255) / 256, 256, 0, stream>>>(x, ret, bsc, undo, out);
}
